// GolemModel_39470749450397
// MI455X (gfx1250) — matrix operand order repaired and checked by reading, not compiled, not run on hardware
//
#include <hip/hip_runtime.h>
#include <hip/hip_bf16.h>

// ---------------- problem constants (from reference) ----------------
#define N_T   2048          // batch
#define E_CNT 1890          // number of (r,c) edge pairs for D=100, DIST=10
#define HID   64
#define NSF   52            // NS_
#define NPF   12            // NP_
#define DD    100
#define M_TILE 256          // n-rows per workgroup (16 waves x 16 rows)
#define BLK    512          // 16 wave32 waves
#define TOLV  1.0e-4f

typedef __attribute__((ext_vector_type(16))) _Float16 v16h;
typedef __attribute__((ext_vector_type(8)))  float    v8f;

// fast sigmoid: v_exp_f32 + v_rcp_f32 (avoids the precise-division expansion)
__device__ __forceinline__ float sigmoidf_(float x) {
    return __builtin_amdgcn_rcpf(1.0f + __expf(-x));
}

// ---------------- zero-fill B (scatter only writes E of D*D per row) ----------------
__global__ __launch_bounds__(256) void golem_zero(float4* __restrict__ p, long n4) {
    long i = (long)blockIdx.x * blockDim.x + threadIdx.x;
    long stride = (long)gridDim.x * blockDim.x;
    float4 z; z.x = 0.f; z.y = 0.f; z.z = 0.f; z.w = 0.f;
    for (; i < n4; i += stride) p[i] = z;
}

// ---------------- fused per-edge MLP + WMMA GEMM + scatter ----------------
__global__ __launch_bounds__(BLK) void golem_main(
    const float* __restrict__ T,   const float* __restrict__ W1,  const float* __restrict__ b1,
    const float* __restrict__ W1p, const float* __restrict__ b1p, const float* __restrict__ W2,
    const float* __restrict__ b2,  const float* __restrict__ W3,  const float* __restrict__ b3,
    const float* __restrict__ beta,const float* __restrict__ bias_p,
    float* __restrict__ outB)
{
    const int e   = blockIdx.y;            // edge index
    const int n0  = blockIdx.x * M_TILE;   // batch tile base
    const int tid = threadIdx.x;
    const int lane = tid & 31;             // wave32
    const int wv   = tid >> 5;             // 16 waves per block

    // LDS: padded rows -> bank spread
    __shared__ __align__(16) _Float16 sH [M_TILE][HID + 4];   // activations (f16)
    __shared__ __align__(16) _Float16 sW2[HID]   [HID + 8];   // W2[e] (f16), row=k_in, col=k_out
    __shared__ float sB2[HID];
    __shared__ float sW3s[HID];
    __shared__ int   sRC[2];

    // --- derive (row, col) of edge e on device (ROWS/COLS are not inputs) ---
    if (tid == 0) {
        int rem = e, r = 0;
        for (;;) {
            int lo  = (r - 10 < 0)  ? 0  : r - 10;
            int hi  = (r + 10 > 99) ? 99 : r + 10;
            int cnt = hi - lo;                       // (+1 cols, -1 for c==r)
            if (rem < cnt) {
                int c = lo + rem;
                if (c >= r) c++;                     // skip diagonal
                sRC[0] = r; sRC[1] = c;
                break;
            }
            rem -= cnt; ++r;
        }
    }

    if (tid < HID) {
        sB2 [tid] = b2[(size_t)e * HID + tid];
        sW3s[tid] = W3[(size_t)e * HID + tid];
    }

    // --- stage W2[e] (64x64) into LDS as f16 (coalesced global reads) ---
    {
        const float* w2e = W2 + (size_t)e * HID * HID;
        #pragma unroll
        for (int i = tid; i < HID * HID; i += BLK)
            sW2[i >> 6][i & 63] = (_Float16)w2e[i];
    }

    // --- build H tile (256 x 64) in LDS: each thread owns one hidden k ---
    {
        const int k = tid & 63;
        float wa, ba, bt = 0.f, bp = 0.f;
        if (k < NSF) { wa = W1 [(size_t)e * NSF + k];         ba = b1 [(size_t)e * NSF + k]; }
        else         { wa = W1p[(size_t)e * NPF + (k - NSF)]; ba = b1p[(size_t)e * NPF + (k - NSF)];
                       bt = beta[e] * (3.14159265358979323f / 10.0f);   // beta*pi/PI_ARG
                       bp = bias_p[e]; }
        for (int nl = tid >> 6; nl < M_TILE; nl += BLK / HID) {
            float t = T[n0 + nl];
            float lin = t * wa + ba;
            float v;
            if (k < NSF) v = sigmoidf_(sigmoidf_(lin));           // sigmoid(sigmoid(x1))
            else         v = sigmoidf_(__cosf(bt * lin + bp));    // sigmoid(cos(...))
            sH[nl][k] = (_Float16)v;
        }
    }
    __syncthreads();

    // --- WMMA: each wave computes 16 rows x 64 cols; K=64 in 2 steps ---
    // A frag layout (ISA 7.12.2, 16-bit A 16x32): M = lane%16;
    //   lanes 0-15 hold K = {0..7,16..23}+base, lanes 16-31 hold K = {8..15,24..31}+base,
    //   element pair j -> K = (j<4 ? 2j : 16+2(j-4)) (+8 for hi lanes).
    // B frag (32x16): N = lane%16, element i -> K same as A (table order, +8 for hi lanes).
    const int mrow  = (wv << 4) + (lane & 15);
    const int khalf = (lane >= 16) ? 8 : 0;

    v8f acc[4] = {};   // 4 N-tiles of 16

    #pragma unroll
    for (int s = 0; s < 2; ++s) {
        union { v16h h; unsigned u[8]; } af;
        const _Float16* hp = &sH[mrow][0];
        const int kb = s * 32 + khalf;
        #pragma unroll
        for (int j = 0; j < 8; ++j) {
            int ko = kb + ((j < 4) ? (2 * j) : (16 + 2 * (j - 4)));
            af.u[j] = *(const unsigned*)&hp[ko];       // two packed f16 (K=ko,ko+1)
        }
        #pragma unroll
        for (int nt = 0; nt < 4; ++nt) {
            v16h bf;
            #pragma unroll
            for (int i = 0; i < 16; ++i)
                bf[i] = sW2[kb + ((i < 8) ? i : (8 + i))][nt * 16 + (lane & 15)];
            acc[nt] = __builtin_amdgcn_wmma_f32_16x16x32_f16(
                false, af.h, false, bf, (short)0, acc[nt], false, false);
        }
    }

    // --- epilogue: out[n] = sum_k sigmoid(acc+b2[k]) * W3[k]  (+b3, threshold) ---
    // C/D layout: element r, lane<16 -> M=r ; lane>=16 -> M=r+8 ; N = nt*16 + lane%16
    float po[8];
    #pragma unroll
    for (int r = 0; r < 8; ++r) po[r] = 0.f;
    #pragma unroll
    for (int nt = 0; nt < 4; ++nt) {
        const int col = nt * 16 + (lane & 15);
        const float bb = sB2[col], w3 = sW3s[col];
        #pragma unroll
        for (int r = 0; r < 8; ++r)
            po[r] += sigmoidf_(acc[nt][r] + bb) * w3;
    }
    #pragma unroll
    for (int r = 0; r < 8; ++r) {                      // reduce 16 lanes (wave32 halves)
        float s = po[r];
        s += __shfl_xor(s, 1, 16);
        s += __shfl_xor(s, 2, 16);
        s += __shfl_xor(s, 4, 16);
        s += __shfl_xor(s, 8, 16);
        po[r] = s;
    }

    if ((lane & 15) == 0) {                            // lanes 0 and 16 hold the sums
        const int rr = sRC[0], cc = sRC[1];
        const float bb3 = b3[e];
        const int rowbase = (wv << 4) + ((lane >= 16) ? 8 : 0);
        #pragma unroll
        for (int r = 0; r < 8; ++r) {
            float o = po[r] + bb3;
            if (fabsf(o) < TOLV) o = 0.f;
            size_t gn = (size_t)(n0 + rowbase + r);
            outB[gn * (size_t)(DD * DD) + (size_t)rr * DD + cc] = o;
        }
    }
}

extern "C" void kernel_launch(void* const* d_in, const int* in_sizes, int n_in,
                              void* d_out, int out_size, void* d_ws, size_t ws_size,
                              hipStream_t stream) {
    (void)in_sizes; (void)n_in; (void)d_ws; (void)ws_size; (void)out_size;
    const float* T     = (const float*)d_in[0];
    const float* W1    = (const float*)d_in[1];
    const float* b1    = (const float*)d_in[2];
    const float* W1p   = (const float*)d_in[3];
    const float* b1p   = (const float*)d_in[4];
    const float* W2    = (const float*)d_in[5];
    const float* b2    = (const float*)d_in[6];
    const float* W3    = (const float*)d_in[7];
    const float* b3    = (const float*)d_in[8];
    const float* beta  = (const float*)d_in[9];
    const float* biasp = (const float*)d_in[10];
    float* outB = (float*)d_out;

    // 1) zero B (N*D*D floats = 5,120,000 float4, exact)
    const long n4 = (long)N_T * DD * DD / 4;
    golem_zero<<<2048, 256, 0, stream>>>((float4*)outB, n4);

    // 2) fused MLP + WMMA GEMM + scatter: grid = (n-tiles, edges)
    dim3 grid(N_T / M_TILE, E_CNT);
    golem_main<<<grid, BLK, 0, stream>>>(T, W1, b1, W1p, b1p, W2, b2, W3, b3, beta, biasp, outB);
}
